// VocoderNetwork_55233279426991
// MI455X (gfx1250) — matrix operand order repaired and checked by reading, not compiled, not run on hardware
//
#include <hip/hip_runtime.h>

// ---------------------------------------------------------------------------
// Types / WMMA helpers (CDNA5 gfx1250, wave32)
// ---------------------------------------------------------------------------
typedef _Float16 v16h __attribute__((ext_vector_type(16)));
typedef _Float16 v8h  __attribute__((ext_vector_type(8)));
typedef float    v8f  __attribute__((ext_vector_type(8)));

#define DEVFN __device__ __forceinline__

DEVFN v8f wmma32f16(v16h a, v16h b, v8f c) {
  // D = A(16x32 f16) * B(32x16 f16) + C(16x16 f32)
  return __builtin_amdgcn_wmma_f32_16x16x32_f16(
      /*neg_a=*/false, a, /*neg_b=*/false, b,
      /*c_mod=*/(short)0, c, /*reuse_a=*/false, /*reuse_b=*/false);
}

DEVFN v8f splat8(float v) {
  v8f r;
#pragma unroll
  for (int i = 0; i < 8; ++i) r[i] = v;
  return r;
}

// Branch-free tanh: gfx1250 hardware v_tanh_f32 transcendental when available;
// fall back to a v_exp_f32-based rational form (no EXEC divergence either way).
DEVFN float fast_tanh(float x) {
#if __has_builtin(__builtin_amdgcn_tanhf)
  return __builtin_amdgcn_tanhf(x);
#else
  float e = __expf(2.0f * x);          // v_exp_f32 (fast path, branch-free)
  return 1.0f - 2.0f / (e + 1.0f);     // saturates correctly at +/-1
#endif
}

DEVFN float sigmoid_t(float g) { return 0.5f * fast_tanh(0.5f * g) + 0.5f; }

// A fragment: 16 rows x 32 K, f16, row-major A with compile-time lda (halves).
// ISA layout: lanes 0-15 -> M=lane, K {k0..k0+7, k0+16..k0+23};
//             lanes 16-31 -> M=lane-16, K {k0+8..k0+15, k0+24..k0+31}.
template <unsigned LDA>
DEVFN v16h load_afrag(const _Float16* __restrict__ A, unsigned row0,
                      unsigned k0, unsigned lane) {
  unsigned r  = row0 + (lane & 15u);
  unsigned ko = k0 + ((lane < 16u) ? 0u : 8u);
  const v8h* p = (const v8h*)(A + (size_t)r * LDA + ko);
  v8h lo = p[0];      // K = ko .. ko+7
  v8h hi = p[2];      // K = ko+16 .. ko+23
  v16h a;
#pragma unroll
  for (int i = 0; i < 8; ++i) { a[i] = lo[i]; a[8 + i] = hi[i]; }
  return a;
}

// A fragment gathered per-row from cond[b][32], b = row >> lshift (K=32).
DEVFN v16h load_afrag_cond(const _Float16* __restrict__ C, unsigned row0,
                           unsigned lshift, unsigned lane) {
  unsigned r = row0 + (lane & 15u);
  unsigned b = r >> lshift;
  unsigned ko = (lane < 16u) ? 0u : 8u;
  const v8h* p = (const v8h*)(C + (size_t)b * 32 + ko);
  v8h lo = p[0];
  v8h hi = p[2];
  v16h a;
#pragma unroll
  for (int i = 0; i < 8; ++i) { a[i] = lo[i]; a[8 + i] = hi[i]; }
  return a;
}

// B fragment: 32 K x 16 N from W[k][n], leading dim ldw.
DEVFN v16h load_bfrag(const _Float16* __restrict__ W, unsigned ldw, unsigned k0,
                      unsigned n0, unsigned lane) {
  unsigned n  = n0 + (lane & 15u);
  unsigned kb = k0 + ((lane < 16u) ? 0u : 8u);
  v16h b;
#pragma unroll
  for (int i = 0; i < 16; ++i) b[i] = W[(size_t)(kb + i + ((i >= 8) ? 8u : 0u)) * ldw + n];
  return b;
}

// ---------------------------------------------------------------------------
// Gated WMMA layer: h_out = tanh(a)*sigmoid(g) + r, a/g/r from one GEMM family
//   A rows: m=(b,l_out) -> 64 contiguous halves of h_in (k=2,s=2 conv == K=64)
//   cond contribution folded in as a 3rd WMMA (K=32) for in/gate branches.
// Wpack: [3 branches][96 rows (64 conv + 32 cond)][COUT] f16, bias: [3][COUT].
// COUT compile-time (32 / 256) -> immediate-offset stores, shift-based indexing.
// ---------------------------------------------------------------------------
template <unsigned COUT>
__global__ __launch_bounds__(128)
void gated_layer_kernel(const _Float16* __restrict__ hin,
                        const _Float16* __restrict__ cond,
                        const _Float16* __restrict__ Wpack,
                        const float* __restrict__ bias,
                        _Float16* __restrict__ hout,
                        unsigned M, unsigned lshift) {
  const unsigned lane = threadIdx.x & 31u;
  const unsigned wave = threadIdx.x >> 5;
  const unsigned n0   = blockIdx.y << 4;
  const _Float16* Wi = Wpack;
  const _Float16* Wg = Wpack + 96u * COUT;
  const _Float16* Wr = Wpack + 2u * 96u * COUT;

  // Hoist all weight fragments (weights are tiny; amortized over M loop).
  v16h Bi0 = load_bfrag(Wi, COUT, 0,  n0, lane);
  v16h Bi1 = load_bfrag(Wi, COUT, 32, n0, lane);
  v16h Bic = load_bfrag(Wi, COUT, 64, n0, lane);
  v16h Bg0 = load_bfrag(Wg, COUT, 0,  n0, lane);
  v16h Bg1 = load_bfrag(Wg, COUT, 32, n0, lane);
  v16h Bgc = load_bfrag(Wg, COUT, 64, n0, lane);
  v16h Br0 = load_bfrag(Wr, COUT, 0,  n0, lane);
  v16h Br1 = load_bfrag(Wr, COUT, 32, n0, lane);

  const unsigned col = n0 + (lane & 15u);
  const float bi = bias[0 * COUT + col];
  const float bg = bias[1 * COUT + col];
  const float br = bias[2 * COUT + col];

  const unsigned mtiles = M >> 4;
  for (unsigned mt = blockIdx.x * 4u + wave; mt < mtiles; mt += gridDim.x * 4u) {
    const unsigned row0 = mt << 4;
    v16h A0 = load_afrag<64>(hin, row0, 0,  lane);
    v16h A1 = load_afrag<64>(hin, row0, 32, lane);
    v16h Ac = load_afrag_cond(cond, row0, lshift, lane);

    v8f accA = splat8(bi), accG = splat8(bg), accR = splat8(br);
    accA = wmma32f16(A0, Bi0, accA);
    accA = wmma32f16(A1, Bi1, accA);
    accA = wmma32f16(Ac, Bic, accA);
    accG = wmma32f16(A0, Bg0, accG);
    accG = wmma32f16(A1, Bg1, accG);
    accG = wmma32f16(Ac, Bgc, accG);
    accR = wmma32f16(A0, Br0, accR);
    accR = wmma32f16(A1, Br1, accR);

    const unsigned rbase = row0 + ((lane < 16u) ? 0u : 8u);
    _Float16* p = hout + (size_t)rbase * COUT + col;
#pragma unroll
    for (int i = 0; i < 8; ++i) {
      float hv = fast_tanh(accA[i]) * sigmoid_t(accG[i]) + accR[i];
      p[(size_t)i * COUT] = (_Float16)hv;   // immediate-offset b16 stores
    }
  }
}

// ---------------------------------------------------------------------------
// Generic WMMA GEMM: C[M,N]f32 = A[M,K]f16 * W[K,N]f16 + bias[N]
// All tile dims compile-time. (K=32 layer-0 cond terms; K=256 final proj.)
// ---------------------------------------------------------------------------
template <unsigned KSTEPS, unsigned N, unsigned LDA>
__global__ __launch_bounds__(128)
void gemm_bias_kernel(const _Float16* __restrict__ A,
                      const _Float16* __restrict__ W,
                      const float* __restrict__ bias,
                      float* __restrict__ C, unsigned M) {
  const unsigned lane = threadIdx.x & 31u;
  const unsigned wave = threadIdx.x >> 5;
  const unsigned n0   = blockIdx.y << 4;

  v16h Bf[KSTEPS];
#pragma unroll
  for (unsigned s = 0; s < KSTEPS; ++s) Bf[s] = load_bfrag(W, N, s * 32u, n0, lane);

  const unsigned col = n0 + (lane & 15u);
  const float bb = bias[col];

  const unsigned mtiles = M >> 4;
  for (unsigned mt = blockIdx.x * 4u + wave; mt < mtiles; mt += gridDim.x * 4u) {
    const unsigned row0 = mt << 4;
    v8f acc = splat8(bb);
#pragma unroll
    for (unsigned s = 0; s < KSTEPS; ++s)
      acc = wmma32f16(load_afrag<LDA>(A, row0, s * 32u, lane), Bf[s], acc);
    const unsigned rbase = row0 + ((lane < 16u) ? 0u : 8u);
    float* p = C + (size_t)rbase * N + col;
#pragma unroll
    for (int i = 0; i < 8; ++i)
      p[(size_t)i * N] = acc[i];
  }
}

// ---------------------------------------------------------------------------
// Layer 0: Cin=1, k=2 conv (too skinny for WMMA) + precomputed cond terms.
// hout: [4000][256][32] f16
// ---------------------------------------------------------------------------
__global__ void layer0_kernel(const float* __restrict__ signal,
                              const float* __restrict__ w_in,
                              const float* __restrict__ w_gate,
                              const float* __restrict__ w_res,
                              const float* __restrict__ b_in,
                              const float* __restrict__ b_gate,
                              const float* __restrict__ b_res,
                              const float* __restrict__ cc_in,
                              const float* __restrict__ cc_gate,
                              _Float16* __restrict__ hout) {
  unsigned idx = blockIdx.x * blockDim.x + threadIdx.x;
  if (idx >= 4000u * 256u * 32u) return;
  unsigned o = idx & 31u;
  unsigned l = (idx >> 5) & 255u;
  unsigned b = idx >> 13;
  float s0 = signal[b + 2u * l];
  float s1 = signal[b + 2u * l + 1u];
  float a = w_in[2u * o] * s0 + w_in[2u * o + 1u] * s1 + b_in[o] + cc_in[b * 32u + o];
  float g = w_gate[2u * o] * s0 + w_gate[2u * o + 1u] * s1 + b_gate[o] + cc_gate[b * 32u + o];
  float r = w_res[2u * o] * s0 + w_res[2u * o + 1u] * s1 + b_res[o];
  hout[idx] = (_Float16)(fast_tanh(a) * sigmoid_t(g) + r);
}

// ---------------------------------------------------------------------------
// Conditioning MLP (tiny): [20,60]->[20,120]->[20,6400] -> cond f16 [4000,32]
// ---------------------------------------------------------------------------
__global__ void cond1_kernel(const float* __restrict__ mgc,
                             const float* __restrict__ w1,
                             const float* __restrict__ b1,
                             float* __restrict__ c1) {
  unsigned idx = blockIdx.x * blockDim.x + threadIdx.x;
  if (idx >= 20u * 120u) return;
  unsigned j = idx % 120u, f = idx / 120u;
  float s = b1[j];
#pragma unroll 4
  for (unsigned k = 0; k < 60u; ++k) s += mgc[f * 60u + k] * w1[j * 60u + k];
  c1[idx] = fast_tanh(s);
}

__global__ void cond2_kernel(const float* __restrict__ c1,
                             const float* __restrict__ w2,
                             const float* __restrict__ b2,
                             _Float16* __restrict__ cond) {
  unsigned idx = blockIdx.x * blockDim.x + threadIdx.x;
  if (idx >= 20u * 6400u) return;
  unsigned j = idx % 6400u, f = idx / 6400u;
  float s = b2[j];
#pragma unroll 4
  for (unsigned k = 0; k < 120u; ++k) s += c1[f * 120u + k] * w2[j * 120u + k];
  float v = fast_tanh(s);
  unsigned b = f * 200u + (j >> 5);
  cond[(size_t)b * 32u + (j & 31u)] = (_Float16)v;
}

// ---------------------------------------------------------------------------
// Weight packing: conv (k=2,s=2) weights + cond weights -> f16 [3][96][Cout]
// K row r<64: r = kk*32+c -> w[o][c][kk] ; r>=64: cond weight (zeros for res)
// ---------------------------------------------------------------------------
__global__ void pack_layer_kernel(const float* __restrict__ w_in,
                                  const float* __restrict__ w_gate,
                                  const float* __restrict__ w_res,
                                  const float* __restrict__ w_cin,
                                  const float* __restrict__ w_cgate,
                                  const float* __restrict__ b_in,
                                  const float* __restrict__ b_gate,
                                  const float* __restrict__ b_res,
                                  const float* __restrict__ b_cin,
                                  const float* __restrict__ b_cgate,
                                  _Float16* __restrict__ Wpack,
                                  float* __restrict__ bias, unsigned Cout) {
  unsigned idx = blockIdx.x * blockDim.x + threadIdx.x;
  if (idx >= 3u * 96u * Cout) return;
  unsigned o = idx % Cout;
  unsigned r = (idx / Cout) % 96u;
  unsigned t = idx / (96u * Cout);
  const float* w = (t == 0u) ? w_in : (t == 1u) ? w_gate : w_res;
  float v;
  if (r < 64u) {
    unsigned kk = r >> 5, c = r & 31u;
    v = w[o * 64u + c * 2u + kk];       // [Cout][32][2]
  } else {
    unsigned c = r - 64u;
    v = (t == 0u) ? w_cin[o * 32u + c] : (t == 1u) ? w_cgate[o * 32u + c] : 0.0f;
  }
  Wpack[idx] = (_Float16)v;
  if (r == 0u) {
    bias[t * Cout + o] = (t == 0u) ? b_in[o] + b_cin[o]
                       : (t == 1u) ? b_gate[o] + b_cgate[o]
                                   : b_res[o];
  }
}

// Wdst[k*N+n] = Wsrc[n*K+k]  (transpose [N,K] f32 -> [K,N] f16)
__global__ void pack_nt_kernel(const float* __restrict__ Wsrc,
                               _Float16* __restrict__ Wdst, unsigned K,
                               unsigned N) {
  unsigned idx = blockIdx.x * blockDim.x + threadIdx.x;
  if (idx >= K * N) return;
  unsigned n = idx % N, k = idx / N;
  Wdst[idx] = (_Float16)Wsrc[n * K + k];
}

__global__ void tanh_h2h_kernel(const _Float16* __restrict__ in,
                                _Float16* __restrict__ out, unsigned n) {
  unsigned idx = blockIdx.x * blockDim.x + threadIdx.x;
  if (idx < n) out[idx] = (_Float16)fast_tanh((float)in[idx]);
}

// ---------------------------------------------------------------------------
// Launch
// ---------------------------------------------------------------------------
extern "C" void kernel_launch(void* const* d_in, const int* in_sizes, int n_in,
                              void* d_out, int out_size, void* d_ws,
                              size_t ws_size, hipStream_t stream) {
  (void)in_sizes; (void)n_in; (void)out_size;
  const unsigned B = 4000;

  const float* mgc     = (const float*)d_in[0];
  const float* signal  = (const float*)d_in[1];
  const float* cond_w1 = (const float*)d_in[92];
  const float* cond_b1 = (const float*)d_in[93];
  const float* cond_w2 = (const float*)d_in[94];
  const float* cond_b2 = (const float*)d_in[95];
  const float* sm_w    = (const float*)d_in[96];
  const float* sm_b    = (const float*)d_in[97];
  auto LPF = [&](int l, int j) { return (const float*)d_in[2 + 10 * l + j]; };

  // ---- workspace carve-up (all 256B aligned) ----
  char* ws = (char*)d_ws;
  size_t off = 0;
  auto carve = [&](size_t bytes) -> void* {
    void* p = ws + off;
    off += (bytes + 255) & ~(size_t)255;
    return p;
  };
  _Float16* bufA  = (_Float16*)carve((size_t)B * 256 * 32 * 2);  // 65.5 MB
  _Float16* bufB  = (_Float16*)carve((size_t)B * 128 * 32 * 2);  // 32.8 MB
  _Float16* cond  = (_Float16*)carve((size_t)B * 32 * 2);
  float*    cc_in = (float*)carve((size_t)B * 32 * 4);
  float*    cc_gt = (float*)carve((size_t)B * 32 * 4);
  _Float16* pre   = (_Float16*)carve((size_t)B * 256 * 2);
  float*    c1    = (float*)carve(20 * 120 * 4);
  _Float16* Wsm   = (_Float16*)carve(256 * 256 * 2);
  _Float16* Wcc0  = (_Float16*)carve(32 * 32 * 2);
  _Float16* Wcc1  = (_Float16*)carve(32 * 32 * 2);
  _Float16* Wpack[9]; float* Lbias[9];
  for (int l = 1; l <= 8; ++l) {
    unsigned Cout = (l == 8) ? 256 : 32;
    Wpack[l] = (_Float16*)carve((size_t)3 * 96 * Cout * 2);
    Lbias[l] = (float*)carve((size_t)3 * Cout * 4);
  }
  if (off > ws_size) return;  // insufficient workspace

  // ---- 1. weight packing ----
  for (int l = 1; l <= 8; ++l) {
    unsigned Cout = (l == 8) ? 256 : 32;
    unsigned tot = 3 * 96 * Cout;
    pack_layer_kernel<<<(tot + 255) / 256, 256, 0, stream>>>(
        LPF(l, 0), LPF(l, 1), LPF(l, 2), LPF(l, 6), LPF(l, 7),
        LPF(l, 3), LPF(l, 4), LPF(l, 5), LPF(l, 8), LPF(l, 9),
        Wpack[l], Lbias[l], Cout);
  }
  pack_nt_kernel<<<(256 * 256 + 255) / 256, 256, 0, stream>>>(sm_w, Wsm, 256, 256);
  pack_nt_kernel<<<(32 * 32 + 255) / 256, 256, 0, stream>>>(LPF(0, 6), Wcc0, 32, 32);
  pack_nt_kernel<<<(32 * 32 + 255) / 256, 256, 0, stream>>>(LPF(0, 7), Wcc1, 32, 32);

  // ---- 2. conditioning MLP ----
  cond1_kernel<<<(2400 + 127) / 128, 128, 0, stream>>>(mgc, cond_w1, cond_b1, c1);
  cond2_kernel<<<(128000 + 127) / 128, 128, 0, stream>>>(c1, cond_w2, cond_b2, cond);

  // ---- 3. layer-0 cond contributions via WMMA GEMM (K=32) ----
  {
    dim3 grid((250 + 3) / 4, 32 / 16);
    gemm_bias_kernel<1, 32, 32><<<grid, 128, 0, stream>>>(cond, Wcc0, LPF(0, 8), cc_in, B);
    gemm_bias_kernel<1, 32, 32><<<grid, 128, 0, stream>>>(cond, Wcc1, LPF(0, 9), cc_gt, B);
  }

  // ---- 4. layer 0 (Cin=1) ----
  {
    unsigned tot = B * 256 * 32;
    layer0_kernel<<<(tot + 255) / 256, 256, 0, stream>>>(
        signal, LPF(0, 0), LPF(0, 1), LPF(0, 2), LPF(0, 3), LPF(0, 4),
        LPF(0, 5), cc_in, cc_gt, bufA);
  }

  // ---- 5. gated WMMA layers 1..8, ping-pong ----
  _Float16* cur = bufA;
  _Float16* nxt = bufB;
  unsigned Lout = 128;
  for (int l = 1; l <= 8; ++l) {
    unsigned M = B * Lout;
    unsigned mtiles = M >> 4;
    unsigned gx = (mtiles + 3) / 4;
    if (gx > 2048) gx = 2048;
    if (l == 8) {
      dim3 grid(gx, 256 / 16);
      gated_layer_kernel<256><<<grid, 128, 0, stream>>>(cur, cond, Wpack[l],
                                                        Lbias[l], nxt, M, 8 - l);
    } else {
      dim3 grid(gx, 32 / 16);
      gated_layer_kernel<32><<<grid, 128, 0, stream>>>(cur, cond, Wpack[l],
                                                       Lbias[l], nxt, M, 8 - l);
    }
    _Float16* t = cur; cur = nxt; nxt = t;
    Lout >>= 1;
  }
  // h9 (f16 [4000,256]) is now in `cur`

  // ---- 6. pre = tanh(h9) ----
  tanh_h2h_kernel<<<(B * 256 + 255) / 256, 256, 0, stream>>>(cur, pre, B * 256);

  // ---- 7. final projection: out = pre @ sm_w^T + sm_b (K=256 WMMA GEMM) ----
  {
    dim3 grid((250 + 3) / 4, 256 / 16);
    gemm_bias_kernel<8, 256, 256><<<grid, 128, 0, stream>>>(pre, Wsm, sm_b,
                                                            (float*)d_out, B);
  }
}
